// ColorImplicitNetwork_21766894256740
// MI455X (gfx1250) — compile-verified, matrix operand order checked by reading, not run on hardware
//
#include <hip/hip_runtime.h>
#include <hip/hip_bf16.h>
#include <math.h>

// ---------------- types ----------------
typedef __attribute__((ext_vector_type(16))) _Float16 v16h;
typedef __attribute__((ext_vector_type(8)))  _Float16 v8h;
typedef __attribute__((ext_vector_type(8)))  float    v8f;

// builtin parameter type (per hipcc diagnostic): int __vector(4) with addrspace quals
typedef int v4i __attribute__((vector_size(16)));
typedef __attribute__((address_space(1))) v4i glb_v4i;   // printed as "__device__"
typedef __attribute__((address_space(3))) v4i lds_v4i;   // "__shared__"

#define NUM_LEVELS 16
#define LOG2_T     19
#define TBL        (1u << LOG2_T)
#define N_POINTS   262144
#define ROWS_PER_WG 256          // 8 waves x 32 rows (2 M-tiles per wave)
#define THREADS     256

// floor(16 * 2^(7l/15)) for l=0..15 (matches reference RESOLUTIONS)
__constant__ int c_res[NUM_LEVELS] = {
    16, 22, 30, 42, 58, 80, 111, 153, 212, 294, 406, 561, 776, 1072, 1481, 2048
};

__device__ __forceinline__ unsigned umin_u(unsigned a, unsigned b) { return a < b ? a : b; }

// ---------------- async global->LDS weight staging ----------------
__device__ __forceinline__ void async_copy_to_lds(_Float16* __restrict__ dst_lds,
                                                  const _Float16* __restrict__ src_glb,
                                                  int bytes, int tid) {
#if __has_builtin(__builtin_amdgcn_global_load_async_to_lds_b128)
    char* l = (char*)dst_lds;
    char* g = (char*)src_glb;
#pragma unroll 4
    for (int ofs = tid * 16; ofs < bytes; ofs += THREADS * 16) {
        __builtin_amdgcn_global_load_async_to_lds_b128(
            (glb_v4i*)(g + ofs), (lds_v4i*)(l + ofs), 0, 0);
    }
#else
#pragma unroll 4
    for (int ofs = tid * 16; ofs < bytes; ofs += THREADS * 16)
        *(float4*)((char*)dst_lds + ofs) = *(const float4*)((const char*)src_glb + ofs);
#endif
}

__device__ __forceinline__ void wait_async_lds() {
#if __has_builtin(__builtin_amdgcn_s_wait_asynccnt)
    __builtin_amdgcn_s_wait_asynccnt(0);
#elif defined(__gfx1250__)
    asm volatile("s_wait_asynccnt 0" ::: "memory");
#endif
}

// Load one 16x32 f16 A/B-fragment (wave32 layout):
// per lane, halves 0..7  = row[k0 .. k0+7], halves 8..15 = row[k0+16 .. k0+23]
// with k0 = kBlockBase + 8*laneHi folded in by the caller.
__device__ __forceinline__ v16h frag16(const _Float16* __restrict__ rowp, int k0) {
    v8h c0 = *(const v8h*)(rowp + k0);
    v8h c1 = *(const v8h*)(rowp + k0 + 16);
    v16h r;
#pragma unroll
    for (int i = 0; i < 8; ++i) { r[i] = c0[i]; r[i + 8] = c1[i]; }
    return r;
}

// One 256->256 layer on the wave's 32 rows (2 M-tiles), B-fragments from LDS.
// H: LDS f16 [256][256]; Wl: LDS f16 [256][256] (pre-transposed Wt[n][k]).
__device__ __forceinline__ void layer256_lds(_Float16* __restrict__ H,
                                             const _Float16* __restrict__ Wl,
                                             const float* __restrict__ bias,
                                             int w, int lane) {
    const int lm = lane & 15;
    const int hi = lane >> 4;
    const int base = 32 * w;

    // Register-resident A fragments for both M-tiles, all 8 K-steps.
    v16h a[2][8];
#pragma unroll
    for (int m = 0; m < 2; ++m) {
        const _Float16* arow = H + (base + 16 * m + lm) * 256;
#pragma unroll
        for (int kt = 0; kt < 8; ++kt)
            a[m][kt] = frag16(arow, 32 * kt + 8 * hi);
    }

#pragma unroll
    for (int nt = 0; nt < 16; ++nt) {
        v8f acc0 = {0.f, 0.f, 0.f, 0.f, 0.f, 0.f, 0.f, 0.f};
        v8f acc1 = {0.f, 0.f, 0.f, 0.f, 0.f, 0.f, 0.f, 0.f};
        const _Float16* wrow = Wl + (16 * nt + lm) * 256;
#pragma unroll
        for (int kt = 0; kt < 8; ++kt) {
            v16h b = frag16(wrow, 32 * kt + 8 * hi);   // loaded once, used twice
            acc0 = __builtin_amdgcn_wmma_f32_16x16x32_f16(
                false, a[0][kt], false, b, (short)0, acc0, false, false);
            acc1 = __builtin_amdgcn_wmma_f32_16x16x32_f16(
                false, a[1][kt], false, b, (short)0, acc1, false, false);
        }
        const int col = 16 * nt + lm;
        const float bv = bias[col];
#pragma unroll
        for (int r = 0; r < 8; ++r) {
            float v0 = acc0[r] + bv; v0 = v0 > 0.f ? v0 : 0.f;
            float v1 = acc1[r] + bv; v1 = v1 > 0.f ? v1 : 0.f;
            H[(base +      r + 8 * hi) * 256 + col] = (_Float16)v0;
            H[(base + 16 + r + 8 * hi) * 256 + col] = (_Float16)v1;
        }
    }
}

__global__ __launch_bounds__(THREADS)
void ngp_mlp_fused(const float* __restrict__ input,        // [N,3]
                   const int*   __restrict__ obj_idx,      // [N]
                   const float* __restrict__ table,        // [16, 2^19, 2]
                   const float* __restrict__ emb,          // [64, 32]
                   const _Float16* __restrict__ W1t,       // [256][64]  f16
                   const float* __restrict__ b1,
                   const _Float16* __restrict__ W2t,       // [256][256] f16
                   const float* __restrict__ b2,
                   const _Float16* __restrict__ W3t,       // [256][256] f16
                   const float* __restrict__ b3,
                   const float* __restrict__ W4,           // [256,3] f32
                   const float* __restrict__ b4,           // [3]
                   float* __restrict__ out)                // [N,3]
{
    __shared__ __align__(16) _Float16 H[ROWS_PER_WG * 256];   // 128 KB activations
    __shared__ __align__(16) _Float16 WB[256 * 256];          // 128 KB weight buffer

    const int tid  = threadIdx.x;
    const int w    = tid >> 5;          // wave id 0..7
    const int lane = tid & 31;
    const int lm   = lane & 15;
    const int hi   = lane >> 4;

    // Kick off async W1 staging; it drains while we do the hash gathers.
    async_copy_to_lds(WB, W1t, 256 * 64 * 2, tid);

    // ---------------- Phase 1: hash-grid encode + embedding -> H[row][0..63] ----
    const int rowL = 32 * w + lane;                   // local row 0..255 (1 lane per row)
    const int gp   = blockIdx.x * ROWS_PER_WG + rowL;

    const float x = input[3 * gp + 0];
    const float y = input[3 * gp + 1];
    const float z = input[3 * gp + 2];
    const float px = x * (1.f / 3.f) + 0.5f;          // ((x/1.5)+1)*0.5
    const float py = y * (1.f / 3.f) + 0.5f;
    const float pz = z * (1.f / 3.f) + 0.5f;

#pragma unroll
    for (int lev = 0; lev < NUM_LEVELS; ++lev) {
        const unsigned res = (unsigned)c_res[lev];
        const unsigned rm  = res - 1u;
        const float rs = (float)rm;

        const float fx = px * rs, fy = py * rs, fz = pz * rs;
        const float ix = floorf(fx), iy = floorf(fy), iz = floorf(fz);
        const float tx = fx - ix, ty = fy - iy, tz = fz - iz;
        const unsigned cx0 = (unsigned)ix, cy0 = (unsigned)iy, cz0 = (unsigned)iz;

        float f0 = 0.f, f1 = 0.f;
#pragma unroll
        for (int c = 0; c < 8; ++c) {
            const unsigned ox = (c >> 2) & 1u, oy = (c >> 1) & 1u, oz = c & 1u;
            const unsigned cx = umin_u(cx0 + ox, rm);
            const unsigned cy = umin_u(cy0 + oy, rm);
            const unsigned cz = umin_u(cz0 + oz, rm);
            unsigned h = cx ^ (cy * 2654435761u) ^ (cz * 805459861u);
            h &= (TBL - 1u);
            const float* fp = table + ((unsigned)(lev << LOG2_T) + h) * 2u;
            const float wx = ox ? tx : 1.f - tx;
            const float wy = oy ? ty : 1.f - ty;
            const float wz = oz ? tz : 1.f - tz;
            const float wt = wx * wy * wz;
            f0 += wt * fp[0];
            f1 += wt * fp[1];
        }
        H[rowL * 256 + 2 * lev + 0] = (_Float16)f0;
        H[rowL * 256 + 2 * lev + 1] = (_Float16)f1;
    }

    {   // object embedding -> k 32..63
        const int ob = obj_idx[gp];
        const float* ep = emb + ob * 32;
#pragma unroll
        for (int j = 0; j < 32; ++j)
            H[rowL * 256 + 32 + j] = (_Float16)ep[j];
    }

    wait_async_lds();           // W1 staged
    __syncthreads();

    // ---------------- Layer 1: 64 -> 256 (2 K-steps), W1 from LDS ----------------
    {
        const int base = 32 * w;
        v16h a[2][2];
#pragma unroll
        for (int m = 0; m < 2; ++m) {
            const _Float16* arow = H + (base + 16 * m + lm) * 256;
            a[m][0] = frag16(arow, 0 + 8 * hi);
            a[m][1] = frag16(arow, 32 + 8 * hi);
        }
#pragma unroll
        for (int nt = 0; nt < 16; ++nt) {
            v8f acc0 = {0.f, 0.f, 0.f, 0.f, 0.f, 0.f, 0.f, 0.f};
            v8f acc1 = {0.f, 0.f, 0.f, 0.f, 0.f, 0.f, 0.f, 0.f};
            const _Float16* wrow = WB + (16 * nt + lm) * 64;
#pragma unroll
            for (int kt = 0; kt < 2; ++kt) {
                v16h b = frag16(wrow, 32 * kt + 8 * hi);
                acc0 = __builtin_amdgcn_wmma_f32_16x16x32_f16(
                    false, a[0][kt], false, b, (short)0, acc0, false, false);
                acc1 = __builtin_amdgcn_wmma_f32_16x16x32_f16(
                    false, a[1][kt], false, b, (short)0, acc1, false, false);
            }
            const int col = 16 * nt + lm;
            const float bv = b1[col];
#pragma unroll
            for (int r = 0; r < 8; ++r) {
                float v0 = acc0[r] + bv; v0 = v0 > 0.f ? v0 : 0.f;
                float v1 = acc1[r] + bv; v1 = v1 > 0.f ? v1 : 0.f;
                H[(base +      r + 8 * hi) * 256 + col] = (_Float16)v0;
                H[(base + 16 + r + 8 * hi) * 256 + col] = (_Float16)v1;
            }
        }
    }

    // ---------------- Layer 2: swap W2 into LDS, compute ----------------
    __syncthreads();                                  // everyone done reading W1
    async_copy_to_lds(WB, W2t, 256 * 256 * 2, tid);
    wait_async_lds();
    __syncthreads();
    layer256_lds(H, WB, b2, w, lane);

    // ---------------- Layer 3: swap W3 into LDS, compute ----------------
    __syncthreads();
    async_copy_to_lds(WB, W3t, 256 * 256 * 2, tid);
    wait_async_lds();
    __syncthreads();
    layer256_lds(H, WB, b3, w, lane);

    // ---------------- Layer 4: 256 -> 3 + sigmoid (one lane per row) --------
    {
        float a0 = 0.f, a1 = 0.f, a2 = 0.f;
        const _Float16* hp = H + rowL * 256;
#pragma unroll 8
        for (int k = 0; k < 256; ++k) {
            const float hv = (float)hp[k];
            a0 += hv * W4[3 * k + 0];
            a1 += hv * W4[3 * k + 1];
            a2 += hv * W4[3 * k + 2];
        }
        const float v0 = a0 + b4[0];
        const float v1 = a1 + b4[1];
        const float v2 = a2 + b4[2];
        out[3 * gp + 0] = 1.f / (1.f + expf(-v0));
        out[3 * gp + 1] = 1.f / (1.f + expf(-v1));
        out[3 * gp + 2] = 1.f / (1.f + expf(-v2));
    }
}

// Convert + transpose weights into f16:  Wt[n][k] = (f16) W[k][n]
__global__ __launch_bounds__(256)
void prep_weights(const float* __restrict__ W1, const float* __restrict__ W2,
                  const float* __restrict__ W3,
                  _Float16* __restrict__ W1t, _Float16* __restrict__ W2t,
                  _Float16* __restrict__ W3t)
{
    const int n = blockIdx.x;    // output column 0..255
    const int t = threadIdx.x;   // input row (K)
    if (t < 64) W1t[n * 64 + t] = (_Float16)W1[t * 256 + n];
    W2t[n * 256 + t] = (_Float16)W2[t * 256 + n];
    W3t[n * 256 + t] = (_Float16)W3[t * 256 + n];
}

extern "C" void kernel_launch(void* const* d_in, const int* in_sizes, int n_in,
                              void* d_out, int out_size, void* d_ws, size_t ws_size,
                              hipStream_t stream) {
    const float* input    = (const float*)d_in[0];
    const int*   obj_idx  = (const int*)  d_in[1];
    const float* table    = (const float*)d_in[2];
    const float* emb      = (const float*)d_in[3];
    const float* W1 = (const float*)d_in[4];  const float* b1 = (const float*)d_in[5];
    const float* W2 = (const float*)d_in[6];  const float* b2 = (const float*)d_in[7];
    const float* W3 = (const float*)d_in[8];  const float* b3 = (const float*)d_in[9];
    const float* W4 = (const float*)d_in[10]; const float* b4 = (const float*)d_in[11];
    float* out = (float*)d_out;

    char* ws = (char*)d_ws;
    _Float16* W1t = (_Float16*)(ws);                          // 32 KB
    _Float16* W2t = (_Float16*)(ws + 32 * 1024);              // 128 KB
    _Float16* W3t = (_Float16*)(ws + 32 * 1024 + 128 * 1024); // 128 KB

    prep_weights<<<256, 256, 0, stream>>>(W1, W2, W3, W1t, W2t, W3t);

    ngp_mlp_fused<<<N_POINTS / ROWS_PER_WG, THREADS, 0, stream>>>(
        input, obj_idx, table, emb,
        W1t, b1, W2t, b2, W3t, b3, W4, b4, out);
}
